// FastRCNNPredictorPaired_14542759264794
// MI455X (gfx1250) — matrix operand order repaired and checked by reading, not compiled, not run on hardware
//
#include <hip/hip_runtime.h>
#include <hip/hip_bf16.h>

typedef __attribute__((ext_vector_type(16))) _Float16 v16h;
typedef __attribute__((ext_vector_type(8)))  _Float16 v8h;
typedef __attribute__((ext_vector_type(4)))  _Float16 v4h;
typedef __attribute__((ext_vector_type(8)))  float    v8f;

#define DD 1024
#define PP 300
#define BB 64
#define NBOX 604          // (NUM_CLASSES+1)*4
#define NCOMB 608         // 604 bbox cols + Wa + Wb + W1 + 1 pad, = 38*16
#define NTILES 38
#define MROWS 64          // rows of x per block (4 M-subtiles of 16)
#define LDA 1032          // padded LDS row stride in halfs: 516 dwords % 64 banks = 4

// ---------------------------------------------------------------------------
// K0: build combined weight matrix, transposed + converted to f16.
//   n <  604 : W_bbox[k, n]; 604: Wa; 605: Wb; 606: W1; 607: pad(0)
// ---------------------------------------------------------------------------
__global__ __launch_bounds__(256) void k0_build_wcomb(
    const float* __restrict__ W_bbox, const float* __restrict__ W_pair,
    const float* __restrict__ W_cls, _Float16* __restrict__ WcombT)
{
    const int n = blockIdx.x;
    for (int k = threadIdx.x; k < DD; k += 256) {
        float v;
        if (n < NBOX)        v = W_bbox[(size_t)k * NBOX + n];
        else if (n == 604)   v = W_pair[k];
        else if (n == 605)   v = W_pair[DD + k];
        else if (n == 606)   v = W_cls[k];
        else                 v = 0.0f;
        WcombT[(size_t)n * DD + k] = (_Float16)v;
    }
}

// ---------------------------------------------------------------------------
// K1: v = W_theta @ Wc (1024-vector), c0 = b_theta·Wc + b_pair.
// ---------------------------------------------------------------------------
__global__ __launch_bounds__(256) void k1_theta_vec(
    const float* __restrict__ W_theta, const float* __restrict__ W_pair,
    const float* __restrict__ b_theta, const float* __restrict__ b_pair,
    float* __restrict__ v, float* __restrict__ c0)
{
    const int w    = blockIdx.x * 8 + (threadIdx.x >> 5);
    const int lane = threadIdx.x & 31;
    const float* __restrict__ Wc = W_pair + 2 * DD;
    if (w < DD) {
        const float* rowp = W_theta + (size_t)w * DD;
        float s = 0.0f;
        for (int j = lane; j < DD; j += 32) s += rowp[j] * Wc[j];
        for (int o = 16; o; o >>= 1) s += __shfl_down(s, o, 32);
        if (lane == 0) v[w] = s;
    } else if (w == DD) {
        float s = 0.0f;
        for (int j = lane; j < DD; j += 32) s += b_theta[j] * Wc[j];
        for (int o = 16; o; o >>= 1) s += __shfl_down(s, o, 32);
        if (lane == 0) *c0 = s + b_pair[0];
    }
}

// ---------------------------------------------------------------------------
// K2: per-batch scalars. One wave per dot.
// ---------------------------------------------------------------------------
__global__ __launch_bounds__(256) void k2_batch_dots(
    const float* __restrict__ relation, const float* __restrict__ subject,
    const float* __restrict__ obj, const float* __restrict__ W_cls,
    const float* __restrict__ b_cls, const float* __restrict__ v,
    const float* __restrict__ c0, float* __restrict__ rel_term,
    float* __restrict__ sb, float* __restrict__ ob)
{
    const int w    = blockIdx.x * 8 + (threadIdx.x >> 5);   // 0..191
    const int lane = threadIdx.x & 31;
    const int kind = w >> 6;
    const int b    = w & 63;
    const float* vecA;
    const float* vecW;
    if (kind == 0)      { vecA = relation + (size_t)b * DD; vecW = v; }
    else if (kind == 1) { vecA = subject  + (size_t)b * DD; vecW = W_cls + DD; }
    else                { vecA = obj      + (size_t)b * DD; vecW = W_cls + DD; }
    float s = 0.0f;
    for (int j = lane; j < DD; j += 32) s += vecA[j] * vecW[j];
    for (int o = 16; o; o >>= 1) s += __shfl_down(s, o, 32);
    if (lane == 0) {
        if (kind == 0)      rel_term[b] = s + *c0;
        else if (kind == 1) sb[b] = s + b_cls[0];
        else                ob[b] = s + b_cls[0];
    }
}

// ---------------------------------------------------------------------------
// K3: fused WMMA GEMM (19200 x 1024) @ (1024 x 608).
// 64-row LDS slab (padded stride -> conflict-free ds_load_b128); per K-step
// one B fragment from L2 feeds FOUR wmmas (4 M-subtiles).
// Manual PING-PONG double buffering across K-steps: buffer y is loaded before
// buffer x is consumed, forcing distinct registers (no coalescing) so every
// wmma group waits only on counters with a full step of newer loads
// outstanding (loadcnt<=2, dscnt<=8) instead of draining to zero.
// ---------------------------------------------------------------------------
#define LOAD_A(d0, d1, kk)                                                    \
    do {                                                                      \
        _Pragma("unroll")                                                     \
        for (int ms = 0; ms < 4; ++ms) {                                      \
            const _Float16* Ap_ = Ab + ms * (16 * LDA) + (kk);                \
            d0[ms] = *(const v8h*)(Ap_);                                      \
            d1[ms] = *(const v8h*)(Ap_ + 16);                                 \
        }                                                                     \
    } while (0)

#define LOAD_B(c0_, c1_, kk)                                                  \
    do {                                                                      \
        c0_ = *(const v8h*)(Bp + (kk));                                       \
        c1_ = *(const v8h*)(Bp + (kk) + 16);                                  \
    } while (0)

#define CONSUME(s0, s1, bb0, bb1)                                             \
    do {                                                                      \
        v16h bf_;                                                             \
        _Pragma("unroll")                                                     \
        for (int t = 0; t < 8; ++t) { bf_[t] = bb0[t]; bf_[t + 8] = bb1[t]; } \
        _Pragma("unroll")                                                     \
        for (int ms = 0; ms < 4; ++ms) {                                      \
            v16h af_;                                                         \
            _Pragma("unroll")                                                 \
            for (int t = 0; t < 8; ++t) {                                     \
                af_[t] = s0[ms][t]; af_[t + 8] = s1[ms][t];                   \
            }                                                                 \
            acc[ms] = __builtin_amdgcn_wmma_f32_16x16x32_f16(                 \
                false, af_, false, bf_, (short)0, acc[ms], false, false);     \
        }                                                                     \
    } while (0)

__global__ __launch_bounds__(256) void k3_gemm_wmma(
    const float* __restrict__ x, const _Float16* __restrict__ WcombT,
    const float* __restrict__ b_bbox, float* __restrict__ out_bbox,
    float* __restrict__ colT, float* __restrict__ rowT, float* __restrict__ xwT)
{
    __shared__ _Float16 sA[MROWS * LDA];      // ~129 KB
    const int tid   = threadIdx.x;
    const int mbase = blockIdx.x * MROWS;

    // Stage + convert the 64x1024 f32 slab to f16 in LDS (padded rows).
    const float4* xg = (const float4*)(x + (size_t)mbase * DD);
    for (int e = tid; e < MROWS * (DD / 4); e += 256) {
        const int r  = e >> 8;                // row 0..63
        const int kc = (e & 255) * 4;         // col 0..1020
        float4 f = xg[e];
        v4h h;
        h[0] = (_Float16)f.x; h[1] = (_Float16)f.y;
        h[2] = (_Float16)f.z; h[3] = (_Float16)f.w;
        *(v4h*)(&sA[r * LDA + kc]) = h;
    }
    __syncthreads();

    const int lane  = tid & 31;
    const int wave  = tid >> 5;
    const int m     = lane & 15;          // A row within 16-row subtile
    const int halfA = lane >> 4;          // A: K sub-offset 0 / 8
    const int nIn   = lane & 15;          // B/D column within tile
    const int kBoff = (lane >> 4) * 8;    // B: K sub-offset 0 / 8

    const _Float16* Ab = &sA[m * LDA + halfA * 8];

    for (int nt = wave; nt < NTILES; nt += 8) {
        v8f acc[4];
        acc[0] = (v8f){}; acc[1] = (v8f){}; acc[2] = (v8f){}; acc[3] = (v8f){};
        const _Float16* Bp = WcombT + (size_t)(nt * 16 + nIn) * DD + kBoff;

        v8h a0x[4], a1x[4], a0y[4], a1y[4];
        v8h b0x, b1x, b0y, b1y;

        // Prologue: buffer x <- k = 0.
        LOAD_B(b0x, b1x, 0);
        LOAD_A(a0x, a1x, 0);

        // Steady state: 15 ping-pong iterations covering k = 0 .. 928.
        for (int k0 = 0; k0 < DD - 64; k0 += 64) {
            LOAD_B(b0y, b1y, k0 + 32);
            LOAD_A(a0y, a1y, k0 + 32);
            CONSUME(a0x, a1x, b0x, b1x);          // step k0 (y outstanding)
            LOAD_B(b0x, b1x, k0 + 64);
            LOAD_A(a0x, a1x, k0 + 64);
            CONSUME(a0y, a1y, b0y, b1y);          // step k0+32 (x outstanding)
        }
        // Tail: x holds k = 960; load + consume k = 992.
        LOAD_B(b0y, b1y, DD - 32);
        LOAD_A(a0y, a1y, DD - 32);
        CONSUME(a0x, a1x, b0x, b1x);              // k = 960
        CONSUME(a0y, a1y, b0y, b1y);              // k = 992

        // D layout: VGPR g -> M = g (lanes 0-15) or g+8 (lanes 16-31), N = nIn.
        const int gcol = nt * 16 + nIn;
        #pragma unroll
        for (int ms = 0; ms < 4; ++ms) {
            #pragma unroll
            for (int g = 0; g < 8; ++g) {
                const int grow = mbase + ms * 16 + ((lane < 16) ? g : g + 8);
                const float val = acc[ms][g];
                if (gcol < NBOX)       out_bbox[(size_t)grow * NBOX + gcol] = val + b_bbox[gcol];
                else if (gcol == 604)  colT[grow] = val;
                else if (gcol == 605)  rowT[grow] = val;
                else if (gcol == 606)  xwT[grow]  = val;
            }
        }
    }
}

// ---------------------------------------------------------------------------
// K4: scores. One block per (b, i) row: 300 pair sigmoids + x_subj/x_obj.
// ---------------------------------------------------------------------------
__global__ __launch_bounds__(256) void k4_scores(
    const float* __restrict__ colT, const float* __restrict__ rowT,
    const float* __restrict__ xwT, const float* __restrict__ rel_term,
    const float* __restrict__ sb, const float* __restrict__ ob,
    float* __restrict__ x_subj, float* __restrict__ x_obj,
    float* __restrict__ pair)
{
    const int bi = blockIdx.x;            // 0..19199
    const int b  = bi / PP;
    const float s_sc = 1.0f / (1.0f + __expf(-(xwT[bi] + sb[b])));
    const float r    = rowT[bi] + rel_term[b];
    for (int j = threadIdx.x; j < PP; j += 256) {
        const int bj = b * PP + j;
        const float o_sc = 1.0f / (1.0f + __expf(-(xwT[bj] + ob[b])));
        const float ps   = 1.0f / (1.0f + __expf(-(r + colT[bj])));
        pair[(size_t)bi * PP + j] = ps * o_sc * s_sc;
    }
    if (threadIdx.x == 0) {
        x_subj[2 * bi]     = 1.0f - s_sc;
        x_subj[2 * bi + 1] = s_sc;
        const float o_self = 1.0f / (1.0f + __expf(-(xwT[bi] + ob[b])));
        x_obj[2 * bi]      = 1.0f - o_self;
        x_obj[2 * bi + 1]  = o_self;
    }
}

// ---------------------------------------------------------------------------
extern "C" void kernel_launch(void* const* d_in, const int* in_sizes, int n_in,
                              void* d_out, int out_size, void* d_ws, size_t ws_size,
                              hipStream_t stream)
{
    (void)in_sizes; (void)n_in; (void)out_size; (void)ws_size;
    const float* x        = (const float*)d_in[0];
    const float* subject  = (const float*)d_in[1];
    const float* obj      = (const float*)d_in[2];
    const float* relation = (const float*)d_in[3];
    const float* W_theta  = (const float*)d_in[4];
    const float* b_theta  = (const float*)d_in[5];
    const float* W_pair   = (const float*)d_in[6];
    const float* b_pair   = (const float*)d_in[7];
    const float* W_cls    = (const float*)d_in[8];
    const float* b_cls    = (const float*)d_in[9];
    const float* W_bbox   = (const float*)d_in[10];
    const float* b_bbox   = (const float*)d_in[11];

    // Workspace layout
    _Float16* WcombT = (_Float16*)d_ws;                         // 608*1024*2 B
    float* fws = (float*)((char*)d_ws + (size_t)NCOMB * DD * 2);
    float* v        = fws;               // 1024
    float* c0       = fws + 1024;        // 1
    float* rel_term = fws + 1025;        // 64
    float* sb       = fws + 1089;        // 64
    float* ob       = fws + 1153;        // 64
    float* colT     = fws + 1217;        // 19200
    float* rowT     = colT + BB * PP;    // 19200
    float* xwT      = rowT + BB * PP;    // 19200

    // Output layout: [x_subj | x_obj | pair_score | bbox_deltas]
    float* out    = (float*)d_out;
    float* x_subj = out;                       // 38400
    float* x_obj  = out + 2 * BB * PP;         // 38400
    float* pair   = out + 4 * BB * PP;         // 5,760,000
    float* bbox   = pair + (size_t)BB * PP * PP;

    k0_build_wcomb<<<NCOMB, 256, 0, stream>>>(W_bbox, W_pair, W_cls, WcombT);
    k1_theta_vec  <<<129,   256, 0, stream>>>(W_theta, W_pair, b_theta, b_pair, v, c0);
    k2_batch_dots <<<24,    256, 0, stream>>>(relation, subject, obj, W_cls, b_cls,
                                              v, c0, rel_term, sb, ob);
    k3_gemm_wmma  <<<(BB * PP) / MROWS, 256, 0, stream>>>(x, WcombT, b_bbox,
                                                          bbox, colT, rowT, xwT);
    k4_scores     <<<BB * PP, 256, 0, stream>>>(colT, rowT, xwT, rel_term, sb, ob,
                                                x_subj, x_obj, pair);
}
